// ElePotentialNet_61383672594919
// MI455X (gfx1250) — matrix operand order repaired and checked by reading, not compiled, not run on hardware
//
#include <hip/hip_runtime.h>
#include <hip/hip_bf16.h>

typedef _Float16 v16h __attribute__((ext_vector_type(16)));
typedef _Float16 v8h  __attribute__((ext_vector_type(8)));
typedef float    v8f  __attribute__((ext_vector_type(8)));

#define C_HID 50
#define INV_SQRT_Z 0.22360679774997896f  /* 1/sqrt(20) */

// Fast SiLU: x * v_rcp_f32(1+exp(-x)) -- avoids the IEEE div_scale/div_fixup chain
// that otherwise dominates the WMMA epilogue VALU cost.
__device__ __forceinline__ float silu(float x) {
  return x * __builtin_amdgcn_rcpf(1.0f + __expf(-x));
}

__device__ __forceinline__ void atomAddF(float* p, float v) {
  __hip_atomic_fetch_add(p, v, __ATOMIC_RELAXED, __HIP_MEMORY_SCOPE_AGENT);
}

__device__ __forceinline__ v8f wmma16(v16h a, v16h b, v8f c) {
  // v_wmma_f32_16x16x32_f16: D = A(16x32 f16) * B(32x16 f16) + C(16x16 f32)
  return __builtin_amdgcn_wmma_f32_16x16x32_f16(false, a, false, b, (short)0, c, false, false);
}

// Rebuild a 16-bit A fragment (16x32, K-chunk kt) from a row-major [16][64] f16 LDS tile.
// Contiguous K order (same as B packing): lane L holds row M=L%16; K = kb..kb+7 (elems 0-7) and kb+8..kb+15 (elems 8-15),
// kb = kt*32 + (L/16)*16.
__device__ __forceinline__ v16h afrag(const _Float16* h, int lane, int kt) {
  int row = lane & 15;
  int kb  = kt * 32 + ((lane >> 4) << 4);
  v8h lo = *(const v8h*)(h + row * 64 + kb);
  v8h hi = *(const v8h*)(h + row * 64 + kb + 8);
  return __builtin_shufflevector(lo, hi, 0,1,2,3,4,5,6,7,8,9,10,11,12,13,14,15);
}

// ---------------- weight prep: f32 -> f16 B-fragment layout ----------------
// B frag (32x16): lane L holds column N = nt*16 + L%16, element j holds K = kt*32 + (L/16)*16 + j.
// wr1f: [3][nt:4][512]   (K=8 padded to 32)
// wr2f: [3][kt:2][nt:4][512]
// wr3f: [3][kt:2][nt:16][512] (N=250 padded to 256)
__global__ void prep_weights(const float* __restrict__ Wr1, const float* __restrict__ Wr2,
                             const float* __restrict__ Wr3,
                             _Float16* __restrict__ wr1f, _Float16* __restrict__ wr2f,
                             _Float16* __restrict__ wr3f) {
  int idx = blockIdx.x * blockDim.x + threadIdx.x;
  if (idx < 6144) {
    int l = idx / 2048, r = idx % 2048;
    int nt = r / 512, q = r % 512, lane = q / 16, j = q % 16;
    int n = nt * 16 + (lane & 15);
    int k = ((lane >> 4) * 16) + j;
    float v = (k < 8) ? Wr1[l * 512 + k * 64 + n] : 0.0f;
    wr1f[idx] = (_Float16)v;
  } else if (idx < 6144 + 12288) {
    int t = idx - 6144;
    int l = t / 4096, r = t % 4096;
    int kt = r / 2048, nt = (r % 2048) / 512, q = r % 512, lane = q / 16, j = q % 16;
    int n = nt * 16 + (lane & 15);
    int k = kt * 32 + ((lane >> 4) * 16) + j;
    wr2f[t] = (_Float16)Wr2[l * 4096 + k * 64 + n];
  } else if (idx < 6144 + 12288 + 49152) {
    int t = idx - 6144 - 12288;
    int l = t / 16384, r = t % 16384;
    int kt = r / 8192, nt = (r % 8192) / 512, q = r % 512, lane = q / 16, j = q % 16;
    int n = nt * 16 + (lane & 15);
    int k = kt * 32 + ((lane >> 4) * 16) + j;
    float v = (n < 250) ? Wr3[l * 16000 + k * 250 + n] : 0.0f;
    wr3f[t] = (_Float16)v;
  }
}

// ---------------- edge geometry + radial basis ----------------
__global__ void geom_kernel(const float* __restrict__ pos, const float* __restrict__ shift,
                            const float* __restrict__ cell, const int* __restrict__ img,
                            const int* __restrict__ esrc, const int* __restrict__ edst,
                            float* __restrict__ Y1, _Float16* __restrict__ radh, int E) {
  int e = blockIdx.x * blockDim.x + threadIdx.x;
  if (e >= E) return;
  int si = esrc[e], di = edst[e];
  const float* Cg = cell + img[si] * 9;
  float sx = shift[e * 3 + 0], sy = shift[e * 3 + 1], sz = shift[e * 3 + 2];
  float ex = pos[di * 3 + 0] - pos[si * 3 + 0] + sx * Cg[0] + sy * Cg[3] + sz * Cg[6];
  float ey = pos[di * 3 + 1] - pos[si * 3 + 1] + sx * Cg[1] + sy * Cg[4] + sz * Cg[7];
  float ez = pos[di * 3 + 2] - pos[si * 3 + 2] + sx * Cg[2] + sy * Cg[5] + sz * Cg[8];
  float r  = sqrtf(ex * ex + ey * ey + ez * ez);
  float rs = fmaxf(r, 1e-9f);
  float inv = __builtin_amdgcn_rcpf(rs);
  const float SQ3 = 1.7320508075688772f;
  Y1[e * 3 + 0] = SQ3 * ex * inv;
  Y1[e * 3 + 1] = SQ3 * ey * inv;
  Y1[e * 3 + 2] = SQ3 * ez * inv;
  float x = r * 0.25f;               // r / R_MAX
  float u = 0.0f;
  if (x < 1.0f) {
    float x2 = x * x, x3 = x2 * x, x6 = x3 * x3;
    u = 1.0f - 28.0f * x6 + 48.0f * x6 * x - 21.0f * x6 * x2;  // p=6 polynomial cutoff
  }
  const float PI_  = 3.14159265358979323846f;
  const float SQ05 = 0.7071067811865476f;  // sqrt(2/R_MAX)
  float base = PI_ * rs * 0.25f;
  float coef = SQ05 * inv * u;
#pragma unroll
  for (int n = 1; n <= 8; ++n)
    radh[e * 8 + (n - 1)] = (_Float16)(__sinf((float)n * base) * coef);
}

// ---------------- node embedding ----------------
__global__ void embed_kernel(const float* __restrict__ W_embed, const int* __restrict__ atype,
                             float* __restrict__ s, int total) {
  int i = blockIdx.x * blockDim.x + threadIdx.x;
  if (i >= total) return;
  int n = i / C_HID, c = i - n * C_HID;
  s[i] = W_embed[atype[n] * C_HID + c];
}

// ---------------- main WMMA edge kernel: radial MLP + message + scatter ----------------
__global__ __launch_bounds__(32) void edge_msg_kernel(
    const _Float16* __restrict__ radh, const float* __restrict__ Y1,
    const float* __restrict__ s, const float* __restrict__ v,
    const int* __restrict__ esrc, const int* __restrict__ edst,
    const _Float16* __restrict__ w1f, const _Float16* __restrict__ w2f,
    const _Float16* __restrict__ w3f,
    float* __restrict__ agg_s, float* __restrict__ agg_v, int E) {
  __shared__ _Float16 h1[16 * 64];
  __shared__ _Float16 h2[16 * 64];
  __shared__ float    wbuf[16 * 256];
  __shared__ float    shY[16][3];
  __shared__ int      shS[16], shD[16];

  const int lane = threadIdx.x;       // 0..31 (one wave per block)
  const int row  = lane & 15;
  const int hlf  = lane >> 4;
  const int e0   = blockIdx.x * 16;

  if (lane < 16) {
    int ge = e0 + lane;
    if (ge < E) {
      shY[lane][0] = Y1[ge * 3 + 0];
      shY[lane][1] = Y1[ge * 3 + 1];
      shY[lane][2] = Y1[ge * 3 + 2];
      shS[lane] = esrc[ge];
      shD[lane] = edst[ge];
    } else {
      shY[lane][0] = shY[lane][1] = shY[lane][2] = 0.0f;
      shS[lane] = 0; shD[lane] = 0;
    }
  }

  // A fragment from radial basis: rows = 16 edges, K=8 zero-padded to 32.
  v16h Ar;
#pragma unroll
  for (int i = 0; i < 16; ++i) Ar[i] = (_Float16)0;
  if (hlf == 0) {
    int ge = e0 + row;
    if (ge < E) {
      v8h r8 = *(const v8h*)(radh + ge * 8);
#pragma unroll
      for (int i = 0; i < 8; ++i) Ar[i] = r8[i];
    }
  }

  // ---- GEMM1: [16,8] @ [8,64] -> silu -> h1 ----
#pragma unroll
  for (int nt = 0; nt < 4; ++nt) {
    v16h B = *(const v16h*)(w1f + nt * 512 + lane * 16);
    v8f acc;
#pragma unroll
    for (int i = 0; i < 8; ++i) acc[i] = 0.0f;
    acc = wmma16(Ar, B, acc);
#pragma unroll
    for (int r = 0; r < 8; ++r)
      h1[(hlf * 8 + r) * 64 + nt * 16 + row] = (_Float16)silu(acc[r]);
  }

  // ---- GEMM2: [16,64] @ [64,64] -> silu -> h2 ----
  v16h A0 = afrag(h1, lane, 0);
  v16h A1 = afrag(h1, lane, 1);
#pragma unroll
  for (int nt = 0; nt < 4; ++nt) {
    v8f acc;
#pragma unroll
    for (int i = 0; i < 8; ++i) acc[i] = 0.0f;
    acc = wmma16(A0, *(const v16h*)(w2f + nt * 512 + lane * 16), acc);
    acc = wmma16(A1, *(const v16h*)(w2f + (4 + nt) * 512 + lane * 16), acc);
#pragma unroll
    for (int r = 0; r < 8; ++r)
      h2[(hlf * 8 + r) * 64 + nt * 16 + row] = (_Float16)silu(acc[r]);
  }

  // ---- GEMM3: [16,64] @ [64,256] -> wbuf (first 250 cols = 5 x 50 path weights) ----
  v16h A2 = afrag(h2, lane, 0);
  v16h A3 = afrag(h2, lane, 1);
#pragma unroll 4
  for (int nt = 0; nt < 16; ++nt) {
    v8f acc;
#pragma unroll
    for (int i = 0; i < 8; ++i) acc[i] = 0.0f;
    acc = wmma16(A2, *(const v16h*)(w3f + nt * 512 + lane * 16), acc);
    acc = wmma16(A3, *(const v16h*)(w3f + (16 + nt) * 512 + lane * 16), acc);
#pragma unroll
    for (int r = 0; r < 8; ++r)
      wbuf[(hlf * 8 + r) * 256 + nt * 16 + row] = acc[r];
  }

  // ---- message formation + atomic scatter: 16 edges x 50 channels ----
#pragma unroll 1
  for (int it = 0; it < 25; ++it) {
    int idx = it * 32 + lane;        // 0..799
    int e = idx / C_HID, c = idx - e * C_HID;
    int ge = e0 + e;
    if (ge >= E) continue;
    int src = shS[e], dst = shD[e];
    float y0 = shY[e][0], y1v = shY[e][1], y2 = shY[e][2];
    float w1 = wbuf[e * 256 + c];
    float w2 = wbuf[e * 256 + 50 + c];
    float w3 = wbuf[e * 256 + 100 + c];
    float w4 = wbuf[e * 256 + 150 + c];
    float w5 = wbuf[e * 256 + 200 + c];
    float ss = s[src * C_HID + c];
    const float* vp = v + (size_t)(src * C_HID + c) * 3;
    float vx = vp[0], vy = vp[1], vz = vp[2];
    atomAddF(&agg_s[dst * 2 * C_HID + c], w1 * ss);
    atomAddF(&agg_s[dst * 2 * C_HID + C_HID + c], w2 * (vx * y0 + vy * y1v + vz * y2));
    float cx = vy * y2 - vz * y1v;
    float cy = vz * y0 - vx * y2;
    float cz = vx * y1v - vy * y0;
    float* ap = agg_v + (size_t)(dst * C_HID + c) * 3;
    atomAddF(ap + 0, w3 * ss * y0 + w4 * vx + w5 * cx);
    atomAddF(ap + 1, w3 * ss * y1v + w4 * vy + w5 * cy);
    atomAddF(ap + 2, w3 * ss * y2 + w4 * vz + w5 * cz);
  }
}

// ---------------- node update (mix + self-connection + gated resnet) ----------------
__global__ __launch_bounds__(128) void node_kernel(
    float* __restrict__ s, float* __restrict__ v,
    const float* __restrict__ aggs, const float* __restrict__ aggv,
    const float* __restrict__ Wmix_s, const float* __restrict__ Wmix_v,
    const float* __restrict__ Wsc_s, const float* __restrict__ Wsc_v,
    const int* __restrict__ atype) {
  __shared__ float sh_as[100], sh_av[150], sh_s[50], sh_v[150], sh_hs[100];
  int n = blockIdx.x, t = threadIdx.x;
  for (int i = t; i < 100; i += 128) sh_as[i] = aggs[n * 100 + i] * INV_SQRT_Z;
  for (int i = t; i < 150; i += 128) sh_av[i] = aggv[n * 150 + i] * INV_SQRT_Z;
  for (int i = t; i < 50;  i += 128) sh_s[i]  = s[n * 50 + i];
  for (int i = t; i < 150; i += 128) sh_v[i]  = v[n * 150 + i];
  __syncthreads();
  if (t < 100) {
    float acc = 0.0f;
    for (int j = 0; j < 100; ++j) acc += sh_as[j] * Wmix_s[j * 100 + t];
    sh_hs[t] = acc;
  }
  __syncthreads();
  int ty = atype[n];
  const float* Ws  = Wsc_s + ty * 2500;
  const float* Wvv = Wsc_v + ty * 2500;
  if (t < 50) {
    int d = t;
    float sc = 0.0f;
    for (int c = 0; c < 50; ++c) sc += sh_s[c] * Ws[c * 50 + d];
    float sg = silu(sh_hs[50 + d]);
    s[n * 50 + d] = sh_s[d] + silu(sh_hs[d] + sc);
    for (int k = 0; k < 3; ++k) {
      float hv = 0.0f, scv = 0.0f;
      for (int c = 0; c < 50; ++c) {
        hv  += sh_av[c * 3 + k] * Wmix_v[c * 50 + d];
        scv += sh_v[c * 3 + k] * Wvv[c * 50 + d];
      }
      v[(size_t)(n * 50 + d) * 3 + k] = sh_v[d * 3 + k] + (hv + scv) * sg;
    }
  }
}

// ---------------- node_feat = s @ Wout, per-graph sum / sumsq / count ----------------
__global__ __launch_bounds__(32) void outfeat_kernel(const float* __restrict__ s,
                                                     const float* __restrict__ Wout,
                                                     const int* __restrict__ img,
                                                     float* gsum, float* gsum2, float* gcnt) {
  __shared__ float sh[50];
  int n = blockIdx.x, f = threadIdx.x;
  for (int i = f; i < 50; i += 32) sh[i] = s[n * 50 + i];
  __syncthreads();
  float acc = 0.0f;
  for (int c = 0; c < 50; ++c) acc += sh[c] * Wout[c * 32 + f];
  int g = img[n];
  atomAddF(&gsum[g * 32 + f], acc);
  atomAddF(&gsum2[g * 32 + f], acc * acc);
  if (f == 0) atomAddF(&gcnt[g], 1.0f);
}

// ---------------- per-graph heads ----------------
__global__ void head_kernel(const float* __restrict__ gsum, const float* __restrict__ gsum2,
                            const float* __restrict__ gcnt,
                            const float* __restrict__ Wv1, const float* __restrict__ Wv2,
                            const float* __restrict__ Wf1, const float* __restrict__ Wf2,
                            float* __restrict__ out, int G) {
  int g = blockIdx.x * blockDim.x + threadIdx.x;
  if (g >= G) return;
  float cnt = gcnt[g];
  float c1  = fmaxf(cnt, 1.0f);
  float rc1 = __builtin_amdgcn_rcpf(c1);
  float rcm1 = __builtin_amdgcn_rcpf(fmaxf(cnt - 1.0f, 1.0f));
  float mean[32], stdv[32];
  for (int f = 0; f < 32; ++f) {
    float m = gsum[g * 32 + f] * rc1;
    mean[f] = m;
    float var = fmaxf(gsum2[g * 32 + f] - c1 * m * m, 0.0f) * rcm1;
    stdv[f] = sqrtf(var + 1e-10f);
  }
  float vac = 0.0f, fer = 0.0f;
  for (int j = 0; j < 16; ++j) {
    float tv = 0.0f, tf = 0.0f;
    for (int f = 0; f < 32; ++f) { tv += mean[f] * Wv1[f * 16 + j]; tf += stdv[f] * Wf1[f * 16 + j]; }
    vac += tv * Wv2[j];
    fer += tf * Wf2[j];
  }
  out[g]     = fer;   // fermi
  out[G + g] = vac;   // vaccum
}

extern "C" void kernel_launch(void* const* d_in, const int* in_sizes, int n_in,
                              void* d_out, int out_size, void* d_ws, size_t ws_size,
                              hipStream_t stream) {
  const float* atom_pos = (const float*)d_in[0];
  const float* eshift   = (const float*)d_in[1];
  const float* cell     = (const float*)d_in[2];
  const int*   atype    = (const int*)d_in[3];
  const int*   esrc     = (const int*)d_in[4];
  const int*   edst     = (const int*)d_in[5];
  const int*   img      = (const int*)d_in[6];
  const float* W_embed  = (const float*)d_in[7];
  const float* Wr1      = (const float*)d_in[8];
  const float* Wr2      = (const float*)d_in[9];
  const float* Wr3      = (const float*)d_in[10];
  const float* Wmix_s   = (const float*)d_in[11];
  const float* Wmix_v   = (const float*)d_in[12];
  const float* Wsc_s    = (const float*)d_in[13];
  const float* Wsc_v    = (const float*)d_in[14];
  const float* Wout     = (const float*)d_in[15];
  const float* Wv1      = (const float*)d_in[16];
  const float* Wv2      = (const float*)d_in[17];
  const float* Wf1      = (const float*)d_in[18];
  const float* Wf2      = (const float*)d_in[19];

  const int N = in_sizes[3];
  const int E = in_sizes[4];
  const int G = in_sizes[2] / 9;

  char* base = (char*)d_ws;
  size_t off = 0;
  auto take = [&](size_t bytes) -> void* {
    void* p = base + off;
    off = (off + bytes + 255) & ~(size_t)255;
    return p;
  };
  float*    Y1   = (float*)   take((size_t)E * 3 * 4);
  _Float16* radh = (_Float16*)take((size_t)E * 8 * 2);
  float*    s_f  = (float*)   take((size_t)N * 50 * 4);
  float*    v_f  = (float*)   take((size_t)N * 150 * 4);
  float*    aggs = (float*)   take((size_t)N * 100 * 4);
  float*    aggv = (float*)   take((size_t)N * 150 * 4);
  _Float16* wr1f = (_Float16*)take(6144 * 2);
  _Float16* wr2f = (_Float16*)take(12288 * 2);
  _Float16* wr3f = (_Float16*)take(49152 * 2);
  float*    gsum  = (float*)  take((size_t)G * 32 * 4);
  float*    gsum2 = (float*)  take((size_t)G * 32 * 4);
  float*    gcnt  = (float*)  take((size_t)G * 4);

  prep_weights<<<(6144 + 12288 + 49152 + 255) / 256, 256, 0, stream>>>(Wr1, Wr2, Wr3,
                                                                       wr1f, wr2f, wr3f);
  geom_kernel<<<(E + 255) / 256, 256, 0, stream>>>(atom_pos, eshift, cell, img, esrc, edst,
                                                   Y1, radh, E);
  embed_kernel<<<(N * 50 + 255) / 256, 256, 0, stream>>>(W_embed, atype, s_f, N * 50);
  (void)hipMemsetAsync(v_f, 0, (size_t)N * 150 * 4, stream);

  for (int l = 0; l < 3; ++l) {
    (void)hipMemsetAsync(aggs, 0, (size_t)N * 100 * 4, stream);
    (void)hipMemsetAsync(aggv, 0, (size_t)N * 150 * 4, stream);
    edge_msg_kernel<<<(E + 15) / 16, 32, 0, stream>>>(radh, Y1, s_f, v_f, esrc, edst,
                                                      wr1f + l * 2048, wr2f + l * 4096,
                                                      wr3f + l * 16384, aggs, aggv, E);
    node_kernel<<<N, 128, 0, stream>>>(s_f, v_f, aggs, aggv,
                                       Wmix_s + l * 10000, Wmix_v + l * 2500,
                                       Wsc_s + l * 5000, Wsc_v + l * 5000, atype);
  }

  (void)hipMemsetAsync(gsum, 0, (size_t)G * 32 * 4, stream);
  (void)hipMemsetAsync(gsum2, 0, (size_t)G * 32 * 4, stream);
  (void)hipMemsetAsync(gcnt, 0, (size_t)G * 4, stream);
  outfeat_kernel<<<N, 32, 0, stream>>>(s_f, Wout, img, gsum, gsum2, gcnt);
  head_kernel<<<(G + 63) / 64, 64, 0, stream>>>(gsum, gsum2, gcnt, Wv1, Wv2, Wf1, Wf2,
                                                (float*)d_out, G);
}
